// Custom_Regressor_29523605192772
// MI455X (gfx1250) — matrix operand order repaired and checked by reading, not compiled, not run on hardware
//
#include <hip/hip_runtime.h>
#include <hip/hip_bf16.h>

typedef __attribute__((ext_vector_type(16))) _Float16 v16h;
typedef __attribute__((ext_vector_type(8)))  float    v8f;

#define NN 100000
#define EE 1600000
#define GG 128
#define NEG_SLOPE 0.01f

__device__ __forceinline__ float lrelu(float v) { return v > 0.f ? v : NEG_SLOPE * v; }

__device__ __forceinline__ float4 addf4(float4 a, float4 b) {
    return make_float4(a.x + b.x, a.y + b.y, a.z + b.z, a.w + b.w);
}

// write A-fragment elements [2*vbase .. 2*vbase+3] from one float4
__device__ __forceinline__ void put4(v16h& a, int vbase, float4 q) {
    a[2 * vbase + 0] = (_Float16)q.x;
    a[2 * vbase + 1] = (_Float16)q.y;
    a[2 * vbase + 2] = (_Float16)q.z;
    a[2 * vbase + 3] = (_Float16)q.w;
}

// ---------------------------------------------------------------------------
// zero scratch regions (agg1, agg2, g are contiguous at start of ws)
// ---------------------------------------------------------------------------
__global__ void zero_kernel(float* __restrict__ p, long n) {
    long i = (long)blockIdx.x * blockDim.x + threadIdx.x;
    long stride = (long)gridDim.x * blockDim.x;
    for (; i < n; i += stride) p[i] = 0.f;
}

// ---------------------------------------------------------------------------
// Pre-swizzle an fp32 weight matrix W[Kreal x C] into WMMA B-fragment order,
// f16, K padded up to nks*32.  One block (1 wave) per (ks, nt) fragment:
// out[blk*256 + lane*8 + v] packs elements (kk,col),(kk+1,col), kk=ks*32+8*half+2*(v&3)+16*(v>>2).
// Conv kernels then load a whole B fragment as one 32B v16h load per lane.
// ---------------------------------------------------------------------------
__global__ void prep_weight(const float* __restrict__ W, int Kreal, int C,
                            unsigned* __restrict__ out) {
    const int lane = threadIdx.x & 31;
    const int ntiles = C >> 4;
    const int blk = blockIdx.x;              // blk = ks*ntiles + nt
    const int nt = blk % ntiles, ks = blk / ntiles;
    const int col = nt * 16 + (lane & 15);
    const int half = lane >> 4;
    unsigned* o = out + (size_t)blk * 256 + lane * 8;
#pragma unroll
    for (int v = 0; v < 8; ++v) {
        int kk = ks * 32 + 8 * half + 2 * (v & 3) + 16 * (v >> 2);
        float f0 = (kk < Kreal) ? W[kk * C + col] : 0.f;
        float f1 = (kk + 1 < Kreal) ? W[(kk + 1) * C + col] : 0.f;
        unsigned u0 = (unsigned)__builtin_bit_cast(unsigned short, (_Float16)f0);
        unsigned u1 = (unsigned)__builtin_bit_cast(unsigned short, (_Float16)f1);
        o[v] = u0 | (u1 << 16);
    }
}

// ---------------------------------------------------------------------------
// GINE edge kernel, conv1: e = edge_attr @ We1 + be1 (ED=16 -> C=32),
// m = relu(x[src] + e), atomic scatter-add into agg[dst].
// One wave = 16 edges. A: 16x16 padded to 16x32 f16. 2 WMMAs (N=32).
// Gathers are batched ahead of the atomics so they issue as one load clause.
// ---------------------------------------------------------------------------
__global__ void conv1_edge(const float* __restrict__ ea, const int* __restrict__ srcI,
                           const int* __restrict__ dstI, const float* __restrict__ x,
                           const v16h* __restrict__ fB, const float* __restrict__ be,
                           float* __restrict__ agg) {
    const int lane = threadIdx.x & 31;
    const int e0 = blockIdx.x * 16;
    const int mrow = lane & 15, half = lane >> 4;

    // A fragment: edge row m, K-chunk [8*half, 8*half+8), rest zero-padded
    const float4* ar4 = (const float4*)ea + (unsigned)(e0 + mrow) * 4u + 2u * half;
    float4 q0 = ar4[0], q1 = ar4[1];
    float4 z = make_float4(0.f, 0.f, 0.f, 0.f);
    v16h a;
    put4(a, 0, q0); put4(a, 2, q1); put4(a, 4, z); put4(a, 6, z);

    // per-lane edge endpoints for its 8 C-rows (row = j + 8*half)
    int sidx[8], didx[8];
#pragma unroll
    for (int j = 0; j < 8; ++j) {
        int e = e0 + j + 8 * half;
        sidx[j] = srcI[e];
        didx[j] = dstI[e];
        __builtin_prefetch(x + (unsigned)sidx[j] * 32u, 0, 1);  // warm gather rows
    }

#pragma unroll
    for (int nt = 0; nt < 2; ++nt) {
        int col = nt * 16 + (lane & 15);
        v16h b = fB[nt * 32 + lane];
        v8f c = {};
        c = __builtin_amdgcn_wmma_f32_16x16x32_f16(false, a, false, b, (short)0, c, false, false);
        float bias = be[col];
        // batch the 8 gathers so they clause together
        float xv[8];
#pragma unroll
        for (int j = 0; j < 8; ++j) xv[j] = x[(unsigned)sidx[j] * 32u + (unsigned)col];
#pragma unroll
        for (int j = 0; j < 8; ++j) {
            float msg = xv[j] + c[j] + bias;
            msg = msg > 0.f ? msg : 0.f;  // relu
            atomicAdd(&agg[(unsigned)didx[j] * 32u + (unsigned)col], msg);
        }
    }
}

// ---------------------------------------------------------------------------
// conv1 node MLP: h_in = x + agg (16x32); t = lrelu(h_in@W1a+b1a) (16x32);
// h1 = lrelu(t@W1b + b1b) (16x64). One wave = 16 nodes. LDS-staged transpose.
// ---------------------------------------------------------------------------
__global__ void conv1_node(const float* __restrict__ x, const float* __restrict__ agg,
                           const v16h* __restrict__ fWa, const float* __restrict__ ba,
                           const v16h* __restrict__ fWb, const float* __restrict__ bb,
                           float* __restrict__ h1) {
    __shared__ float t[16 * 32];
    const int lane = threadIdx.x & 31;
    const int n0 = blockIdx.x * 16;
    const int mrow = lane & 15, half = lane >> 4;

    // A = x + agg (16x32): chunks [8*half,+8) and [16+8*half,+8)
    unsigned r4 = (unsigned)(n0 + mrow) * 8u + 2u * half;  // float4 index
    const float4* xr = (const float4*)x;
    const float4* gr = (const float4*)agg;
    v16h a;
    put4(a, 0, addf4(xr[r4],     gr[r4]));
    put4(a, 2, addf4(xr[r4 + 1], gr[r4 + 1]));
    put4(a, 4, addf4(xr[r4 + 4], gr[r4 + 4]));
    put4(a, 6, addf4(xr[r4 + 5], gr[r4 + 5]));

    // GEMM1: 16x32 @ 32x32
#pragma unroll
    for (int nt = 0; nt < 2; ++nt) {
        int col = nt * 16 + (lane & 15);
        v16h b = fWa[nt * 32 + lane];
        v8f c = {};
        c = __builtin_amdgcn_wmma_f32_16x16x32_f16(false, a, false, b, (short)0, c, false, false);
        float bias = ba[col];
#pragma unroll
        for (int j = 0; j < 8; ++j) t[(j + 8 * half) * 32 + col] = lrelu(c[j] + bias);
    }
    __syncthreads();  // single-wave block: cheap LDS ordering

    // A2 from staged t (row stride 32 floats = 8 float4)
    const float4* ts = (const float4*)t;
    int i0 = mrow * 8 + 2 * half;
    v16h a2;
    put4(a2, 0, ts[i0]);     put4(a2, 2, ts[i0 + 1]);
    put4(a2, 4, ts[i0 + 4]); put4(a2, 6, ts[i0 + 5]);

    // GEMM2: 16x32 @ 32x64, trailing lrelu (post-conv activation)
#pragma unroll
    for (int nt = 0; nt < 4; ++nt) {
        int col = nt * 16 + (lane & 15);
        v16h b = fWb[nt * 32 + lane];
        v8f c = {};
        c = __builtin_amdgcn_wmma_f32_16x16x32_f16(false, a2, false, b, (short)0, c, false, false);
        float bias = bb[col];
#pragma unroll
        for (int j = 0; j < 8; ++j) {
            int row = j + 8 * half;
            h1[(unsigned)(n0 + row) * 64u + (unsigned)col] = lrelu(c[j] + bias);
        }
    }
}

// ---------------------------------------------------------------------------
// conv2 edge kernel: e = edge_attr @ We2 + be2 (16 -> 64), m = relu(h1[src]+e),
// atomic scatter-add into agg2[dst]. 4 N-tiles, batched gathers.
// ---------------------------------------------------------------------------
__global__ void conv2_edge(const float* __restrict__ ea, const int* __restrict__ srcI,
                           const int* __restrict__ dstI, const float* __restrict__ h1,
                           const v16h* __restrict__ fB, const float* __restrict__ be,
                           float* __restrict__ agg) {
    const int lane = threadIdx.x & 31;
    const int e0 = blockIdx.x * 16;
    const int mrow = lane & 15, half = lane >> 4;

    const float4* ar4 = (const float4*)ea + (unsigned)(e0 + mrow) * 4u + 2u * half;
    float4 q0 = ar4[0], q1 = ar4[1];
    float4 z = make_float4(0.f, 0.f, 0.f, 0.f);
    v16h a;
    put4(a, 0, q0); put4(a, 2, q1); put4(a, 4, z); put4(a, 6, z);

    int sidx[8], didx[8];
#pragma unroll
    for (int j = 0; j < 8; ++j) {
        int e = e0 + j + 8 * half;
        sidx[j] = srcI[e];
        didx[j] = dstI[e];
        __builtin_prefetch(h1 + (unsigned)sidx[j] * 64u, 0, 1);
    }

#pragma unroll
    for (int nt = 0; nt < 4; ++nt) {
        int col = nt * 16 + (lane & 15);
        v16h b = fB[nt * 32 + lane];
        v8f c = {};
        c = __builtin_amdgcn_wmma_f32_16x16x32_f16(false, a, false, b, (short)0, c, false, false);
        float bias = be[col];
        float xv[8];
#pragma unroll
        for (int j = 0; j < 8; ++j) xv[j] = h1[(unsigned)sidx[j] * 64u + (unsigned)col];
#pragma unroll
        for (int j = 0; j < 8; ++j) {
            float msg = xv[j] + c[j] + bias;
            msg = msg > 0.f ? msg : 0.f;
            atomicAdd(&agg[(unsigned)didx[j] * 64u + (unsigned)col], msg);
        }
    }
}

// ---------------------------------------------------------------------------
// conv2 node MLP with FUSED global_add_pool:
// h_in = h1 + agg2 (16x64); t = lrelu(h_in@W2a+b2a) (16x128);
// h2 = lrelu(t@W2b + b2b) (16x256) is accumulated directly into
// g[batch[node], :] via atomics (h2 is only consumed by the pool).
// ---------------------------------------------------------------------------
__global__ void conv2_node(const float* __restrict__ h1, const float* __restrict__ agg,
                           const v16h* __restrict__ fWa, const float* __restrict__ ba,
                           const v16h* __restrict__ fWb, const float* __restrict__ bb,
                           const int* __restrict__ batch, float* __restrict__ g) {
    __shared__ float t[16 * 128];
    const int lane = threadIdx.x & 31;
    const int n0 = blockIdx.x * 16;
    const int mrow = lane & 15, half = lane >> 4;

    // A fragments for h_in (16x64): two K-slices, row stride 64 floats = 16 float4
    const float4* xr = (const float4*)h1;
    const float4* gr = (const float4*)agg;
    unsigned rb = (unsigned)(n0 + mrow) * 16u;
    v16h a[2];
#pragma unroll
    for (int ks = 0; ks < 2; ++ks) {
        unsigned i0 = rb + (unsigned)(ks * 8 + 2 * half);
        put4(a[ks], 0, addf4(xr[i0],     gr[i0]));
        put4(a[ks], 2, addf4(xr[i0 + 1], gr[i0 + 1]));
        put4(a[ks], 4, addf4(xr[i0 + 4], gr[i0 + 4]));
        put4(a[ks], 6, addf4(xr[i0 + 5], gr[i0 + 5]));
    }

    // graph ids of this tile's 8 rows per lane (for the fused pool)
    int gid[8];
#pragma unroll
    for (int j = 0; j < 8; ++j) gid[j] = batch[n0 + j + 8 * half];

    // GEMM1: 16x64 @ 64x128 -> t, lrelu.  fragment blk = ks*8 + nt
#pragma unroll
    for (int nt = 0; nt < 8; ++nt) {
        int col = nt * 16 + (lane & 15);
        v8f c = {};
#pragma unroll
        for (int ks = 0; ks < 2; ++ks) {
            v16h b = fWa[(ks * 8 + nt) * 32 + lane];
            c = __builtin_amdgcn_wmma_f32_16x16x32_f16(false, a[ks], false, b, (short)0, c, false, false);
        }
        float bias = ba[col];
#pragma unroll
        for (int j = 0; j < 8; ++j) t[(j + 8 * half) * 128 + col] = lrelu(c[j] + bias);
    }
    __syncthreads();

    // A2 fragments from t (row stride 128 floats = 32 float4): four K-slices
    const float4* ts = (const float4*)t;
    v16h a2[4];
#pragma unroll
    for (int ks = 0; ks < 4; ++ks) {
        int i0 = mrow * 32 + ks * 8 + 2 * half;
        put4(a2[ks], 0, ts[i0]);     put4(a2[ks], 2, ts[i0 + 1]);
        put4(a2[ks], 4, ts[i0 + 4]); put4(a2[ks], 6, ts[i0 + 5]);
    }

    // GEMM2: 16x128 @ 128x256, lrelu, fused pool.  fragment blk = ks*16 + nt
#pragma unroll
    for (int nt = 0; nt < 16; ++nt) {
        int col = nt * 16 + (lane & 15);
        v8f c = {};
#pragma unroll
        for (int ks = 0; ks < 4; ++ks) {
            v16h b = fWb[(ks * 16 + nt) * 32 + lane];
            c = __builtin_amdgcn_wmma_f32_16x16x32_f16(false, a2[ks], false, b, (short)0, c, false, false);
        }
        float bias = bb[col];
#pragma unroll
        for (int j = 0; j < 8; ++j) {
            float val = lrelu(c[j] + bias);
            atomicAdd(&g[(unsigned)gid[j] * 256u + (unsigned)col], val);
        }
    }
}

// ---------------------------------------------------------------------------
// head: 256 -> 128 -> 64 -> 32 -> 1, lrelu between. One block per graph.
// ---------------------------------------------------------------------------
__global__ void head_kernel(const float* __restrict__ g,
                            const float* __restrict__ Wf0, const float* __restrict__ bf0,
                            const float* __restrict__ Wf1, const float* __restrict__ bf1,
                            const float* __restrict__ Wf2, const float* __restrict__ bf2,
                            const float* __restrict__ Wr, const float* __restrict__ br,
                            float* __restrict__ out) {
    __shared__ float s0[128];
    __shared__ float s1[64];
    __shared__ float s2[32];
    const int gi = blockIdx.x;
    const int t = threadIdx.x;  // 128 threads
    const float* grp = g + (unsigned)gi * 256u;

    float acc = bf0[t];
    for (int k = 0; k < 256; ++k) acc += grp[k] * Wf0[k * 128 + t];
    s0[t] = lrelu(acc);
    __syncthreads();

    if (t < 64) {
        acc = bf1[t];
        for (int k = 0; k < 128; ++k) acc += s0[k] * Wf1[k * 64 + t];
        s1[t] = lrelu(acc);
    }
    __syncthreads();

    if (t < 32) {
        acc = bf2[t];
        for (int k = 0; k < 64; ++k) acc += s1[k] * Wf2[k * 32 + t];
        s2[t] = lrelu(acc);
    }
    __syncthreads();

    if (t == 0) {
        acc = br[0];
        for (int k = 0; k < 32; ++k) acc += s2[k] * Wr[k];
        out[gi] = acc;
    }
}

// ---------------------------------------------------------------------------
extern "C" void kernel_launch(void* const* d_in, const int* in_sizes, int n_in,
                              void* d_out, int out_size, void* d_ws, size_t ws_size,
                              hipStream_t stream) {
    const float* x   = (const float*)d_in[0];
    const int*   ei  = (const int*)d_in[1];
    const float* ea  = (const float*)d_in[2];
    const int*   bat = (const int*)d_in[3];
    const float* We1 = (const float*)d_in[4];
    const float* be1 = (const float*)d_in[5];
    const float* W1a = (const float*)d_in[6];
    const float* b1a = (const float*)d_in[7];
    const float* W1b = (const float*)d_in[8];
    const float* b1b = (const float*)d_in[9];
    const float* We2 = (const float*)d_in[10];
    const float* be2 = (const float*)d_in[11];
    const float* W2a = (const float*)d_in[12];
    const float* b2a = (const float*)d_in[13];
    const float* W2b = (const float*)d_in[14];
    const float* b2b = (const float*)d_in[15];
    const float* Wf0 = (const float*)d_in[16];
    const float* bf0 = (const float*)d_in[17];
    const float* Wf1 = (const float*)d_in[18];
    const float* bf1 = (const float*)d_in[19];
    const float* Wf2 = (const float*)d_in[20];
    const float* bf2 = (const float*)d_in[21];
    const float* Wr  = (const float*)d_in[22];
    const float* br  = (const float*)d_in[23];

    const int* srcI = ei;        // edge_index[0]
    const int* dstI = ei + EE;   // edge_index[1]

    // workspace layout (floats):
    // [agg1 N*32 | agg2 N*64 | g G*256 | h1 N*64 | wprep 92*256]
    float* ws   = (float*)d_ws;
    float* agg1 = ws;
    float* agg2 = agg1 + (long)NN * 32;
    float* gbuf = agg2 + (long)NN * 64;
    float* h1   = gbuf + (long)GG * 256;
    unsigned* wprep = (unsigned*)(h1 + (long)NN * 64);

    // pre-swizzled f16 B-fragment regions (1 KB per 16-col x 32-K fragment)
    unsigned* pWe1 = wprep + 0  * 256;  // 2 blocks
    unsigned* pW1a = wprep + 2  * 256;  // 2 blocks
    unsigned* pW1b = wprep + 4  * 256;  // 4 blocks
    unsigned* pWe2 = wprep + 8  * 256;  // 4 blocks
    unsigned* pW2a = wprep + 12 * 256;  // 16 blocks
    unsigned* pW2b = wprep + 28 * 256;  // 64 blocks

    const long zeroN = (long)NN * 96 + (long)GG * 256;  // agg1+agg2+g contiguous
    zero_kernel<<<4096, 256, 0, stream>>>(ws, zeroN);

    prep_weight<<<2,  32, 0, stream>>>(We1, 16, 32,  pWe1);
    prep_weight<<<2,  32, 0, stream>>>(W1a, 32, 32,  pW1a);
    prep_weight<<<4,  32, 0, stream>>>(W1b, 32, 64,  pW1b);
    prep_weight<<<4,  32, 0, stream>>>(We2, 16, 64,  pWe2);
    prep_weight<<<16, 32, 0, stream>>>(W2a, 64, 128, pW2a);
    prep_weight<<<64, 32, 0, stream>>>(W2b, 128, 256, pW2b);

    conv1_edge<<<EE / 16, 32, 0, stream>>>(ea, srcI, dstI, x, (const v16h*)pWe1, be1, agg1);
    conv1_node<<<NN / 16, 32, 0, stream>>>(x, agg1, (const v16h*)pW1a, b1a,
                                           (const v16h*)pW1b, b1b, h1);
    conv2_edge<<<EE / 16, 32, 0, stream>>>(ea, srcI, dstI, h1, (const v16h*)pWe2, be2, agg2);
    conv2_node<<<NN / 16, 32, 0, stream>>>(h1, agg2, (const v16h*)pW2a, b2a,
                                           (const v16h*)pW2b, b2b, bat, gbuf);
    head_kernel<<<GG, 128, 0, stream>>>(gbuf, Wf0, bf0, Wf1, bf1, Wf2, bf2, Wr, br,
                                        (float*)d_out);
}
